// TSPRLModel_27900107555498
// MI455X (gfx1250) — matrix operand order repaired and checked by reading, not compiled, not run on hardware
//
#include <hip/hip_runtime.h>
#include <hip/hip_bf16.h>

typedef __attribute__((ext_vector_type(16))) _Float16 v16h;
typedef __attribute__((ext_vector_type(8)))  _Float16 v8h;
typedef __attribute__((ext_vector_type(8)))  float    v8f;

#define B_    64
#define N_    1000
#define P_    1000
#define D_    64
#define NPAD  1024          // padded feature rows (zeros past 999)
#define PT_   63            // p-tiles of 16 (ceil(1000/16))
#define NG_   16            // n-groups of 64 (ceil(1000/64))
#define WPB   8             // waves per block
#define ALPHA 0.8f
#define BETA  0.2f

// Kernel 1: node_features[b,n,:] = points[b,n,:] @ W + bias as f16, rows
// padded to NPAD with zeros so edge-tile WMMA operands read zeros.
// Layout: feat[(b*NPAD + n)*64 + d].  (Regular-temporal stores: this is the
// hot L2-resident array the GEMM re-reads.)
__global__ void tsp_features_f16(const float* __restrict__ points,
                                 const float* __restrict__ W,
                                 const float* __restrict__ bias,
                                 _Float16* __restrict__ feat) {
    int idx = blockIdx.x * blockDim.x + threadIdx.x;   // B_*NPAD*D_ = 4194304
    int b   = idx >> 16;            // NPAD*D_ = 65536
    int rem = idx & 65535;
    int n   = rem >> 6;
    int d   = rem & 63;
    float v = 0.0f;
    if (n < N_) {
        const float* p = points + ((size_t)(b * N_ + n)) * 2;
        v = p[0] * W[d] + p[1] * W[64 + d] + bias[d];
    }
    feat[(size_t)idx] = (_Float16)v;
}

// Kernel 2: one wave32 computes a 16x64 output tile (4 WMMA n-tiles sharing
// one A fragment). 8 x v_wmma_f32_16x16x32_f16 per wave (K=64, 4 tiles).
// Fused epilogue: out = visited ? -inf : 0.8*adj[cur[p],n] + 0.2*sim.
// adj/mask/out use non-temporal hints so the 580MB stream doesn't evict the
// 8.4MB feature array from the 192MB L2.
__global__ void __launch_bounds__(256)
tsp_sim_wmma(const _Float16* __restrict__ feat,
             const long long* __restrict__ cur,
             const float*     __restrict__ adj,
             const unsigned char* __restrict__ mask,
             float* __restrict__ out) {
    int wave = threadIdx.x >> 5;
    int lane = threadIdx.x & 31;
    int t    = blockIdx.x * WPB + wave;        // < 64*63*16 = 64512 (exact grid)

    int b  = t / (PT_ * NG_);
    int r  = t - b * (PT_ * NG_);
    int pt = r >> 4;                           // p-tile index (0..62)
    int ng = r & 15;                           // n-group index (0..15)
    int p0 = pt * 16;
    int nb = ng * 64;

    int l15 = lane & 15;
    int hs  = lane >> 4;                       // lane half select

    // ---- A operand: gathered feature row cur[b, p0+l15] (clamped) ----
    int pa = p0 + l15; if (pa > P_ - 1) pa = P_ - 1;
    int ca = (int)cur[(size_t)b * P_ + pa];    // always in [0, N_)

    const _Float16* arow = feat + ((size_t)b * NPAD + ca) * D_ + hs * 16;
    v16h afrag[2];
#pragma unroll
    for (int kk = 0; kk < 2; ++kk) {
        // A k order used here (B same): lanes<16 hold K=kb+[0..15]; lanes>=16 K=kb+[16..31]
        v8h x0 = *(const v8h*)(arow + kk * 32);
        v8h x1 = *(const v8h*)(arow + kk * 32 + 8);
        afrag[kk] = __builtin_shufflevector(x0, x1,
                    0,1,2,3,4,5,6,7, 8,9,10,11,12,13,14,15);
    }

    // ---- B operand base: feature row (nb + l15); tile j at +j*1024 halves ----
    const _Float16* brow = feat + ((size_t)b * NPAD + nb + l15) * D_ + hs * 16;

    v8f acc[4] = {};
#pragma unroll
    for (int j = 0; j < 4; ++j) {
#pragma unroll
        for (int kk = 0; kk < 2; ++kk) {
            v16h bm = *(const v16h*)(brow + j * 1024 + kk * 32);
            acc[j] = __builtin_amdgcn_wmma_f32_16x16x32_f16(
                         false, afrag[kk], false, bm, (short)0, acc[j],
                         false, false);
        }
    }

    // ---- broadcast cur-row indices for output rows (EXEC still all-1s) ----
    int cps[8];
#pragma unroll
    for (int i = 0; i < 8; ++i)
        cps[i] = __builtin_amdgcn_ds_bpermute((i + hs * 8) << 2, ca);

    // ---- epilogue ----
    int n = nb + l15;                           // base column, always < 1000
    size_t rowbase = ((size_t)b * P_ + p0 + hs * 8) * N_ + n;
    const unsigned char* mrow = mask + rowbase;
    float*               orow = out  + rowbase;
    const float*         adjb = adj + (size_t)b * N_ * N_ + n;
    const float ninf = -__builtin_inff();

    if (pt < PT_ - 1 && ng < NG_ - 1) {
        // interior: all 16 rows / 64 cols valid -> branch-free
#pragma unroll
        for (int i = 0; i < 8; ++i) {
            const float*         ap = adjb + (size_t)cps[i] * N_;
            const unsigned char* mp = mrow + (size_t)i * N_;
            float*               op = orow + (size_t)i * N_;
            float a0 = __builtin_nontemporal_load(ap);
            float a1 = __builtin_nontemporal_load(ap + 16);
            float a2 = __builtin_nontemporal_load(ap + 32);
            float a3 = __builtin_nontemporal_load(ap + 48);
            unsigned char m0 = __builtin_nontemporal_load(mp);
            unsigned char m1 = __builtin_nontemporal_load(mp + 16);
            unsigned char m2 = __builtin_nontemporal_load(mp + 32);
            unsigned char m3 = __builtin_nontemporal_load(mp + 48);
            float v0 = m0 ? ninf : ALPHA * a0 + BETA * acc[0][i];
            float v1 = m1 ? ninf : ALPHA * a1 + BETA * acc[1][i];
            float v2 = m2 ? ninf : ALPHA * a2 + BETA * acc[2][i];
            float v3 = m3 ? ninf : ALPHA * a3 + BETA * acc[3][i];
            __builtin_nontemporal_store(v0, op);
            __builtin_nontemporal_store(v1, op + 16);
            __builtin_nontemporal_store(v2, op + 32);
            __builtin_nontemporal_store(v3, op + 48);
        }
    } else {
        // edge tiles (~8% of waves): guard rows (p) and cols (n)
#pragma unroll
        for (int i = 0; i < 8; ++i) {
            int p = p0 + i + hs * 8;
            if (p >= P_) continue;
            const float*         ap = adjb + (size_t)cps[i] * N_;
            const unsigned char* mp = mrow + (size_t)i * N_;
            float*               op = orow + (size_t)i * N_;
#pragma unroll
            for (int j = 0; j < 4; ++j) {
                if (n + j * 16 < N_) {
                    float av = __builtin_nontemporal_load(ap + j * 16);
                    unsigned char m = __builtin_nontemporal_load(mp + j * 16);
                    float v = m ? ninf : ALPHA * av + BETA * acc[j][i];
                    __builtin_nontemporal_store(v, op + j * 16);
                }
            }
        }
    }
}

extern "C" void kernel_launch(void* const* d_in, const int* in_sizes, int n_in,
                              void* d_out, int out_size, void* d_ws, size_t ws_size,
                              hipStream_t stream) {
    const float*         points = (const float*)d_in[0];
    const long long*     curnd  = (const long long*)d_in[1];     // int64
    const unsigned char* vmask  = (const unsigned char*)d_in[2]; // bool
    const float*         adj    = (const float*)d_in[3];
    const float*         W      = (const float*)d_in[4];
    const float*         bias   = (const float*)d_in[5];
    float*               out    = (float*)d_out;

    _Float16* feat = (_Float16*)d_ws;   // needs 64*1024*64*2 = 8.4 MB

    // B_*NPAD*D_ / 256 = 16384 blocks
    tsp_features_f16<<<16384, 256, 0, stream>>>(points, W, bias, feat);
    // 64*63*16 waves / 8 per block = 8064 blocks (exact)
    tsp_sim_wmma<<<8064, 256, 0, stream>>>(feat, curnd, adj, vmask, out);
}
